// Local2FWLRefine_85323820302754
// MI455X (gfx1250) — matrix operand order repaired and checked by reading, not compiled, not run on hardware
//
#include <hip/hip_runtime.h>

typedef __attribute__((ext_vector_type(16))) _Float16 v16h;
typedef __attribute__((ext_vector_type(8)))  _Float16 v8h;
typedef __attribute__((ext_vector_type(8)))  float    v8f;
typedef __attribute__((ext_vector_type(4)))  float    v4f;

#define HIDDEN 128
#define NRBF   32
#define KIN    865      // 6*128 + 3*32 + 1
#define KP     896      // padded to multiple of 32
#define KSTEPS (KP / 32)

// ---------------------------------------------------------------------------
// Kernel 1: zero accumulator + counter, convert weights to f16 (transposed to
// [N][K] so each WMMA B-fragment lane reads contiguous K).
// ---------------------------------------------------------------------------
__global__ void prep_kernel(const float* __restrict__ w1, const float* __restrict__ w2,
                            const float* __restrict__ wgw, const float* __restrict__ wgt,
                            _Float16* __restrict__ w1h, _Float16* __restrict__ w2h,
                            _Float16* __restrict__ wgwh, _Float16* __restrict__ wgth,
                            float* __restrict__ macc, int* __restrict__ counter, int E2) {
  size_t stride = (size_t)gridDim.x * blockDim.x;
  size_t tid = (size_t)blockIdx.x * blockDim.x + threadIdx.x;
  size_t nM = (size_t)E2 * HIDDEN;
  for (size_t x = tid; x < nM; x += stride) macc[x] = 0.0f;
  for (size_t x = tid; x < (size_t)HIDDEN * KP; x += stride) {
    int n = (int)(x / KP), k = (int)(x % KP);
    w1h[x] = (k < KIN) ? (_Float16)w1[(size_t)k * HIDDEN + n] : (_Float16)0.0f;
  }
  for (size_t x = tid; x < (size_t)HIDDEN * HIDDEN; x += stride) {
    int n = (int)(x / HIDDEN), k = (int)(x % HIDDEN);
    size_t kn = (size_t)k * HIDDEN + n;
    w2h[x]  = (_Float16)w2[kn];
    wgwh[x] = (_Float16)wgw[kn];
    wgth[x] = (_Float16)wgt[kn];
  }
  if (tid == 0) *counter = 0;
}

// ---------------------------------------------------------------------------
// Kernel 2: wedge construction. src1 is sorted by source; e2 keys are sorted
// (searchsorted invariant from the reference). One thread per e1 edge (i->k);
// scan k's outgoing edges, keep (i,j) pairs that exist in e2.
// ---------------------------------------------------------------------------
__global__ void build_wedges(const int* __restrict__ ei1, const int* __restrict__ ei2,
                             const int* __restrict__ nn_ptr, int E1, int E2,
                             int* __restrict__ counter,
                             int* __restrict__ weik, int* __restrict__ wekj,
                             int* __restrict__ weij, int Wmax) {
  int e = blockIdx.x * blockDim.x + threadIdx.x;
  if (e >= E1) return;
  const int* src1 = ei1;
  const int* dst1 = ei1 + E1;
  const int* src2 = ei2;
  const int* dst2 = ei2 + E2;
  long long N = (long long)(*nn_ptr);
  int vi = src1[e], vk = dst1[e];
  if (vi == vk) return;  // drop self-loops
  // lower_bound of vk in sorted src1
  int lo = 0, hi = E1;
  while (lo < hi) { int mid = (lo + hi) >> 1; if (src1[mid] < vk) lo = mid + 1; else hi = mid; }
  for (int f = lo; f < E1 && src1[f] == vk; ++f) {
    int vj = dst1[f];
    if (vj == vi) continue;
    long long key = (long long)vi * N + vj;
    int a = 0, b = E2;
    while (a < b) {
      int mid = (a + b) >> 1;
      long long km = (long long)src2[mid] * N + dst2[mid];
      if (km < key) a = mid + 1; else b = mid;
    }
    if (a < E2 && ((long long)src2[a] * N + dst2[a]) == key) {
      int w = atomicAdd(counter, 1);
      if (w < Wmax) { weik[w] = e; wekj[w] = f; weij[w] = a; }
    }
  }
}

// ---------------------------------------------------------------------------
// Kernel 3: per-16-wedge-tile fused MLP.
//   X[16][896] f16 (LDS)  -> GEMM1 (28 x v_wmma_f32_16x16x32_f16) -> silu
//   H1[16][128] f16 (LDS) -> GEMM2 (4 k-steps) -> atomicAdd scatter into macc
// 128 threads = 4 wave32; each wave owns two 16-wide N-tiles.
// ---------------------------------------------------------------------------
__global__ __launch_bounds__(128) void wedge_mlp(
    const float* __restrict__ t_e2, const float* __restrict__ h,
    const int* __restrict__ ei1, const int* __restrict__ e1e2,
    const float* __restrict__ rbf1, const float* __restrict__ rbf2,
    const float* __restrict__ sph,
    const _Float16* __restrict__ w1h, const _Float16* __restrict__ w2h,
    const float* __restrict__ b1, const float* __restrict__ b2,
    const int* __restrict__ weik, const int* __restrict__ wekj,
    const int* __restrict__ weij, const int* __restrict__ counter,
    float* __restrict__ macc, int E1, int Wmax) {
  __shared__ _Float16 X[16][KP];        // 28 KB
  __shared__ _Float16 H1[16][HIDDEN];   // 4 KB
  __shared__ int s_eij[16];
  __shared__ int s_val[16];

  int W = *counter;
  if (W > Wmax) W = Wmax;
  int base = blockIdx.x * 16;
  if (base >= W) return;

  int tid = threadIdx.x;
  int s = tid >> 3;       // wedge slot 0..15
  int q = tid & 7;        // 8 gather threads per wedge
  int w = base + s;
  bool valid = w < W;

  int eik = 0, ekj = 0, eij = 0;
  const float* p0 = t_e2; const float* p1 = t_e2; const float* p2 = t_e2;
  const float* p3 = h;    const float* p4 = h;    const float* p5 = h;
  if (valid) {
    eik = weik[w]; ekj = wekj[w]; eij = weij[w];
    int vi = ei1[eik], vk = ei1[(size_t)E1 + eik], vj = ei1[(size_t)E1 + ekj];
    p0 = t_e2 + (size_t)e1e2[eik] * HIDDEN;
    p1 = t_e2 + (size_t)e1e2[ekj] * HIDDEN;
    p2 = t_e2 + (size_t)eij * HIDDEN;
    p3 = h + (size_t)vi * HIDDEN;
    p4 = h + (size_t)vk * HIDDEN;
    p5 = h + (size_t)vj * HIDDEN;
  }
  if (q == 0) { s_eij[s] = eij; s_val[s] = valid ? 1 : 0; }

  // 6 x 128 f32 row segments -> f16 LDS (each thread: 16 floats per segment)
#define GATHER_SEG(P, G)                                                  \
  {                                                                        \
    _Float16* dst = &X[s][(G) * HIDDEN + q * 16];                          \
    if (valid) {                                                           \
      const v4f* f4 = (const v4f*)((P) + q * 16);                          \
      _Pragma("unroll") for (int u = 0; u < 4; ++u) {                      \
        v4f v = f4[u];                                                     \
        _Pragma("unroll") for (int t = 0; t < 4; ++t)                      \
          dst[u * 4 + t] = (_Float16)v[t];                                 \
      }                                                                    \
    } else {                                                               \
      _Pragma("unroll") for (int u = 0; u < 16; ++u) dst[u] = (_Float16)0.0f; \
    }                                                                      \
  }
  GATHER_SEG(p0, 0) GATHER_SEG(p1, 1) GATHER_SEG(p2, 2)
  GATHER_SEG(p3, 3) GATHER_SEG(p4, 4) GATHER_SEG(p5, 5)
#undef GATHER_SEG

  // RBF block: 96 columns starting at 768 (12 per thread)
  for (int u = 0; u < 12; ++u) {
    int c = q * 12 + u;
    float val = 0.0f;
    if (valid) {
      if (c < 32)       val = rbf1[(size_t)eik * NRBF + c];
      else if (c < 64)  val = rbf1[(size_t)ekj * NRBF + (c - 32)];
      else              val = rbf2[(size_t)eij * NRBF + (c - 64)];
    }
    X[s][768 + c] = (_Float16)val;
  }
  // c_feat at 864, zero pad 865..895 (4 per thread)
  for (int u = 0; u < 4; ++u) {
    int c = 864 + q * 4 + u;
    float val = 0.0f;
    if (c == 864 && valid) val = sph[(size_t)eik * 3 + 1] * sph[(size_t)ekj * 3 + 1];
    X[s][c] = (_Float16)val;
  }
  __syncthreads();

  int lane = tid & 31, wv = tid >> 5;
  int half = lane >> 4;                 // lane group
  int lrow = lane & 15;                 // M (A) or N (B/C) within tile
  int kbA = half ? 16 : 0;              // A-frag: lanes 16-31 -> K{16..31}
  int kbB = half ? 16 : 0;              // B-frag: lanes 16-31 -> K{16..31}
  int nt0 = wv * 2, nt1 = wv * 2 + 1;   // this wave's N-tiles
  int n0 = nt0 * 16 + lrow, n1 = nt1 * 16 + lrow;

  // ---- GEMM1: X[16,896] @ W1h^T[128,896] ----
  v8f acc0 = {}, acc1 = {};
  for (int ks = 0; ks < KSTEPS; ++ks) {
    int k0 = ks * 32;
    v8h alo = *(const v8h*)&X[lrow][k0 + kbA];
    v8h ahi = *(const v8h*)&X[lrow][k0 + 8 + kbA];
    v16h a = __builtin_shufflevector(alo, ahi, 0,1,2,3,4,5,6,7,8,9,10,11,12,13,14,15);
    v16h bv0 = *(const v16h*)(w1h + (size_t)n0 * KP + k0 + kbB);
    v16h bv1 = *(const v16h*)(w1h + (size_t)n1 * KP + k0 + kbB);
    acc0 = __builtin_amdgcn_wmma_f32_16x16x32_f16(false, a, false, bv0, (short)0, acc0, false, false);
    acc1 = __builtin_amdgcn_wmma_f32_16x16x32_f16(false, a, false, bv1, (short)0, acc1, false, false);
  }
  // bias + silu -> H1 (C layout: VGPR r -> M = r + half*8, N = lrow)
  float bi0 = b1[n0], bi1 = b1[n1];
#pragma unroll
  for (int r = 0; r < 8; ++r) {
    int row = half * 8 + r;
    float x0 = acc0[r] + bi0;
    float x1 = acc1[r] + bi1;
    H1[row][n0] = (_Float16)(x0 / (1.0f + __expf(-x0)));
    H1[row][n1] = (_Float16)(x1 / (1.0f + __expf(-x1)));
  }
  __syncthreads();

  // ---- GEMM2: H1[16,128] @ W2h^T[128,128] ----
  v8f acc2 = {}, acc3 = {};
  for (int ks = 0; ks < 4; ++ks) {
    int k0 = ks * 32;
    v8h alo = *(const v8h*)&H1[lrow][k0 + kbA];
    v8h ahi = *(const v8h*)&H1[lrow][k0 + 8 + kbA];
    v16h a = __builtin_shufflevector(alo, ahi, 0,1,2,3,4,5,6,7,8,9,10,11,12,13,14,15);
    v16h bv0 = *(const v16h*)(w2h + (size_t)n0 * HIDDEN + k0 + kbB);
    v16h bv1 = *(const v16h*)(w2h + (size_t)n1 * HIDDEN + k0 + kbB);
    acc2 = __builtin_amdgcn_wmma_f32_16x16x32_f16(false, a, false, bv0, (short)0, acc2, false, false);
    acc3 = __builtin_amdgcn_wmma_f32_16x16x32_f16(false, a, false, bv1, (short)0, acc3, false, false);
  }
  // bias + segment-sum scatter (f32 global atomics)
  float bo0 = b2[n0], bo1 = b2[n1];
#pragma unroll
  for (int r = 0; r < 8; ++r) {
    int row = half * 8 + r;
    if (s_val[row]) {
      size_t rowoff = (size_t)s_eij[row] * HIDDEN;
      atomicAdd(&macc[rowoff + n0], acc2[r] + bo0);
      atomicAdd(&macc[rowoff + n1], acc3[r] + bo1);
    }
  }
}

// ---------------------------------------------------------------------------
// Kernel 4: gating. out = t + sigmoid(m@Wgw+bgw) * tanh(t@Wgt+bgt).
// One block per 16 rows of E2; both GEMMs on the matrix pipe.
// ---------------------------------------------------------------------------
__global__ __launch_bounds__(128) void gate_kernel(
    const float* __restrict__ t_e2, const float* __restrict__ macc,
    const _Float16* __restrict__ wgwh, const _Float16* __restrict__ wgth,
    const float* __restrict__ bgw, const float* __restrict__ bgt,
    float* __restrict__ out, int E2) {
  __shared__ _Float16 Tm[16][HIDDEN];
  __shared__ _Float16 Mm[16][HIDDEN];
  __shared__ float    Tf[16][HIDDEN];

  int tid = threadIdx.x;
  int s = tid >> 3, q = tid & 7;
  int row = blockIdx.x * 16 + s;
  bool valid = row < E2;
  const float* tp = t_e2 + (size_t)(valid ? row : 0) * HIDDEN;
  const float* mp = macc + (size_t)(valid ? row : 0) * HIDDEN;
#pragma unroll
  for (int u = 0; u < 4; ++u) {
    v4f tv = valid ? ((const v4f*)(tp + q * 16))[u] : (v4f){0.f, 0.f, 0.f, 0.f};
    v4f mv = valid ? ((const v4f*)(mp + q * 16))[u] : (v4f){0.f, 0.f, 0.f, 0.f};
#pragma unroll
    for (int t = 0; t < 4; ++t) {
      int c = q * 16 + u * 4 + t;
      Tf[s][c] = tv[t];
      Tm[s][c] = (_Float16)tv[t];
      Mm[s][c] = (_Float16)mv[t];
    }
  }
  __syncthreads();

  int lane = tid & 31, wv = tid >> 5;
  int half = lane >> 4, lrow = lane & 15;
  int kbA = half ? 16 : 0, kbB = half ? 16 : 0;
  int n0 = (wv * 2) * 16 + lrow, n1 = (wv * 2 + 1) * 16 + lrow;

  v8f ag0 = {}, ag1 = {}, at0 = {}, at1 = {};
  for (int ks = 0; ks < 4; ++ks) {
    int k0 = ks * 32;
    v8h mlo = *(const v8h*)&Mm[lrow][k0 + kbA];
    v8h mhi = *(const v8h*)&Mm[lrow][k0 + 8 + kbA];
    v16h am = __builtin_shufflevector(mlo, mhi, 0,1,2,3,4,5,6,7,8,9,10,11,12,13,14,15);
    v8h tlo = *(const v8h*)&Tm[lrow][k0 + kbA];
    v8h thi = *(const v8h*)&Tm[lrow][k0 + 8 + kbA];
    v16h atv = __builtin_shufflevector(tlo, thi, 0,1,2,3,4,5,6,7,8,9,10,11,12,13,14,15);
    v16h bw0 = *(const v16h*)(wgwh + (size_t)n0 * HIDDEN + k0 + kbB);
    v16h bw1 = *(const v16h*)(wgwh + (size_t)n1 * HIDDEN + k0 + kbB);
    v16h bt0 = *(const v16h*)(wgth + (size_t)n0 * HIDDEN + k0 + kbB);
    v16h bt1 = *(const v16h*)(wgth + (size_t)n1 * HIDDEN + k0 + kbB);
    ag0 = __builtin_amdgcn_wmma_f32_16x16x32_f16(false, am, false, bw0, (short)0, ag0, false, false);
    ag1 = __builtin_amdgcn_wmma_f32_16x16x32_f16(false, am, false, bw1, (short)0, ag1, false, false);
    at0 = __builtin_amdgcn_wmma_f32_16x16x32_f16(false, atv, false, bt0, (short)0, at0, false, false);
    at1 = __builtin_amdgcn_wmma_f32_16x16x32_f16(false, atv, false, bt1, (short)0, at1, false, false);
  }
  float bg0 = bgw[n0], bg1 = bgw[n1], bt0s = bgt[n0], bt1s = bgt[n1];
#pragma unroll
  for (int r = 0; r < 8; ++r) {
    int srow = half * 8 + r;
    int grow = blockIdx.x * 16 + srow;
    if (grow < E2) {
      float g0 = 1.0f / (1.0f + __expf(-(ag0[r] + bg0)));
      float g1 = 1.0f / (1.0f + __expf(-(ag1[r] + bg1)));
      float h0 = tanhf(at0[r] + bt0s);
      float h1 = tanhf(at1[r] + bt1s);
      out[(size_t)grow * HIDDEN + n0] = Tf[srow][n0] + g0 * h0;
      out[(size_t)grow * HIDDEN + n1] = Tf[srow][n1] + g1 * h1;
    }
  }
}

// ---------------------------------------------------------------------------
extern "C" void kernel_launch(void* const* d_in, const int* in_sizes, int n_in,
                              void* d_out, int out_size, void* d_ws, size_t ws_size,
                              hipStream_t stream) {
  const float* t_e2 = (const float*)d_in[0];
  const float* h    = (const float*)d_in[1];
  const int*   ei1  = (const int*)d_in[2];
  const int*   ei2  = (const int*)d_in[3];
  const int*   e1e2 = (const int*)d_in[4];
  const float* rbf1 = (const float*)d_in[7];
  const float* rbf2 = (const float*)d_in[8];
  const float* sph  = (const float*)d_in[9];
  const int*   nn   = (const int*)d_in[10];
  const float* w1   = (const float*)d_in[11];
  const float* b1   = (const float*)d_in[12];
  const float* w2   = (const float*)d_in[13];
  const float* b2   = (const float*)d_in[14];
  const float* wgw  = (const float*)d_in[15];
  const float* bgw  = (const float*)d_in[16];
  const float* wgt  = (const float*)d_in[17];
  const float* bgt  = (const float*)d_in[18];
  float* out = (float*)d_out;

  int E2 = in_sizes[0] / HIDDEN;
  int E1 = in_sizes[5];
  int N  = in_sizes[1] / HIDDEN;
  long long deg = (N > 0) ? (long long)E1 / N : 1; if (deg < 1) deg = 1;
  long long Wmax = (long long)E1 * deg;  // upper bound on wedge count

  // workspace layout
  char* ws = (char*)d_ws;
  size_t off = 0;
  auto alignup = [](size_t v, size_t a) { return (v + a - 1) & ~(a - 1); };
  float* macc = (float*)(ws + off);      off += (size_t)E2 * HIDDEN * sizeof(float); off = alignup(off, 256);
  _Float16* w1h  = (_Float16*)(ws + off); off += (size_t)HIDDEN * KP * sizeof(_Float16);     off = alignup(off, 256);
  _Float16* w2h  = (_Float16*)(ws + off); off += (size_t)HIDDEN * HIDDEN * sizeof(_Float16); off = alignup(off, 256);
  _Float16* wgwh = (_Float16*)(ws + off); off += (size_t)HIDDEN * HIDDEN * sizeof(_Float16); off = alignup(off, 256);
  _Float16* wgth = (_Float16*)(ws + off); off += (size_t)HIDDEN * HIDDEN * sizeof(_Float16); off = alignup(off, 256);
  int* counter = (int*)(ws + off);       off += 256;
  size_t avail = (ws_size > off) ? (ws_size - off) : 0;
  long long wcap = (long long)(avail / (3 * sizeof(int)));
  if (Wmax > wcap) Wmax = wcap;
  if (Wmax < 1) Wmax = 1;
  int* weik = (int*)(ws + off);          off += (size_t)Wmax * sizeof(int);
  int* wekj = (int*)(ws + off);          off += (size_t)Wmax * sizeof(int);
  int* weij = (int*)(ws + off);

  prep_kernel<<<2048, 256, 0, stream>>>(w1, w2, wgw, wgt, w1h, w2h, wgwh, wgth,
                                        macc, counter, E2);
  build_wedges<<<(E1 + 255) / 256, 256, 0, stream>>>(ei1, ei2, nn, E1, E2, counter,
                                                     weik, wekj, weij, (int)Wmax);
  int wblocks = (int)((Wmax + 15) / 16); if (wblocks < 1) wblocks = 1;
  wedge_mlp<<<wblocks, 128, 0, stream>>>(t_e2, h, ei1, e1e2, rbf1, rbf2, sph,
                                         w1h, w2h, b1, b2, weik, wekj, weij,
                                         counter, macc, E1, (int)Wmax);
  gate_kernel<<<(E2 + 15) / 16, 128, 0, stream>>>(t_e2, macc, wgwh, wgth, bgw, bgt,
                                                  out, E2);
}
